// RNNWithSigmoid_54425825575651
// MI455X (gfx1250) — matrix operand order repaired and checked by reading, not compiled, not run on hardware
//
#include <hip/hip_runtime.h>

// ---------------------------------------------------------------------------
// Problem constants
// ---------------------------------------------------------------------------
#define EMBED 128
#define BATCH 32
#define NSTEP 128
#define GDIM  50257
#define MROWS 4096          // BATCH * NSTEP
#define NCHUNK 64           // N-chunks over vocab
#define CHUNK  832          // 13 * 64 ; 64*832 = 53248 >= 50257
#define VPAD   53248
#define KSTR   136          // bf16 LDS row stride (128 + 8 pad -> bank rotate)
#define ZSTRF  132          // f32 LDS row stride

typedef __attribute__((ext_vector_type(16))) __bf16 v16bf;
typedef __attribute__((ext_vector_type(8)))  float  v8f;

__device__ __forceinline__ unsigned short f2bf(float f) {
    unsigned u = __float_as_uint(f);
    unsigned r = (u + 0x7FFFu + ((u >> 16) & 1u)) >> 16;   // RNE
    return (unsigned short)r;
}
__device__ __forceinline__ float bf2f(unsigned short h) {
    return __uint_as_float(((unsigned)h) << 16);
}

// Hardware tanh (V_TANH_F32) when the builtin exists; branch-free fallback
// (exp overflow/underflow saturates to +-1 correctly, no EXEC divergence).
__device__ __forceinline__ float fast_tanh(float x) {
#if __has_builtin(__builtin_amdgcn_tanhf)
    return __builtin_amdgcn_tanhf(x);
#else
    float e = __expf(2.0f * x);
    return 1.0f - 2.0f / (e + 1.0f);
#endif
}

__device__ __forceinline__ void wait_async0() {
#if __has_builtin(__builtin_amdgcn_s_wait_asynccnt)
    __builtin_amdgcn_s_wait_asynccnt(0);
#else
    asm volatile("s_wait_asynccnt 0x0" ::: "memory");
#endif
}

struct FragBits { unsigned u[8]; };

// A-matrix 16x32 bf16 fragment from row-major [row][k] LDS tile (ISA 7.12.2):
// lane m=l&15, hi=l>>4 ; VGPR j<4: K = 2j+8hi ; j>=4: K = 16+2(j-4)+8hi
__device__ __forceinline__ v16bf ld_a_frag(const unsigned short* lds, int row,
                                           int stride, int kc, int lane) {
    int m = lane & 15, hi = lane >> 4;
    FragBits f;
#pragma unroll
    for (int j = 0; j < 4; ++j) {
        int k = kc + 2 * j + 8 * hi;
        f.u[j] = *(const unsigned*)(lds + (row + m) * stride + k);
    }
#pragma unroll
    for (int j = 4; j < 8; ++j) {
        int k = kc + 16 + 2 * (j - 4) + 8 * hi;
        f.u[j] = *(const unsigned*)(lds + (row + m) * stride + k);
    }
    return __builtin_bit_cast(v16bf, f);
}

// B-matrix 32x16 bf16 fragment; B[k,n] = W[n,k] so tile is row-major [n][k].
// VGPR j<4: K = 2j+8hi ; j>=4: K = 16+2(j-4)+8hi (same as A) ; N = lane&15
__device__ __forceinline__ v16bf ld_b_frag(const unsigned short* lds, int row,
                                           int stride, int kc, int lane) {
    int n = lane & 15, hi = lane >> 4;
    FragBits f;
#pragma unroll
    for (int j = 0; j < 8; ++j) {
        int k = kc + 2 * (j & 3) + 8 * hi + 16 * (j >> 2);
        f.u[j] = *(const unsigned*)(lds + (row + n) * stride + k);
    }
    return __builtin_bit_cast(v16bf, f);
}

// Async DMA of a 64-row x 128-bf16 tile (256B/row, contiguous rows in global)
// into an LDS tile with KSTR padding. 256 threads x 4 b128 transfers, tracked
// by ASYNCcnt -- no VGPR round-trip, overlaps with WMMA compute.
__device__ __forceinline__ void load_tile_async(unsigned short* dst,
                                                const unsigned short* src,
                                                int tid) {
#pragma unroll
    for (int i = 0; i < 4; ++i) {
        int idx = tid + i * 256;                 // 0..1023
        int r = idx >> 4, c = (idx & 15) * 16;   // row, byte col
        unsigned ldsoff = (unsigned)(size_t)(dst + r * KSTR) + (unsigned)c;
        unsigned voff   = (unsigned)(r * 256 + c);
        asm volatile("global_load_async_to_lds_b128 %0, %1, %2"
                     :: "v"(ldsoff), "v"(voff), "s"(src) : "memory");
    }
}

// ---------------------------------------------------------------------------
// Kernel 0: convert Wv (f32, 50257x128) -> bf16, zero-padded to 53248 rows
// ---------------------------------------------------------------------------
__global__ __launch_bounds__(256) void k_cvt(const float* __restrict__ Wv,
                                             unsigned short* __restrict__ wvb) {
    int i = blockIdx.x * 256 + threadIdx.x;
    if (i < VPAD * EMBED)
        wvb[i] = (i < GDIM * EMBED) ? f2bf(Wv[i]) : (unsigned short)0;
}

// ---------------------------------------------------------------------------
// Kernel 1: recurrence. One workgroup, 16 waves, each owns a 16x16 tile of
// the [32 x 128] state. Wt^T fragments preloaded once into registers; per
// step: emit z (bf16) to zs, C-init = z+bt (f32), 4x WMMA bf16, tanh, store.
// ---------------------------------------------------------------------------
__global__ __launch_bounds__(512) void k_rnn(const int* __restrict__ zi,
                                             const float* __restrict__ latent,
                                             const float* __restrict__ Wt,
                                             const float* __restrict__ bt,
                                             unsigned short* __restrict__ zs) {
    __shared__ float          zf[BATCH * ZSTRF];
    __shared__ unsigned short zb[BATCH * KSTR];
    __shared__ unsigned short wt[EMBED * KSTR];
    __shared__ float          btl[EMBED];

    int tid = threadIdx.x, lane = tid & 31, w = tid >> 5;

    for (int i = tid; i < BATCH * EMBED; i += 512) {
        int b = i >> 7, e = i & 127;
        float v = latent[(size_t)zi[b] * EMBED + e];
        zf[b * ZSTRF + e] = v;
        zb[b * KSTR + e]  = f2bf(v);
    }
    for (int i = tid; i < EMBED * EMBED; i += 512) {
        int r = i >> 7, c = i & 127;
        wt[r * KSTR + c] = f2bf(Wt[r * EMBED + c]);   // row n, col k (== Wt^T as B)
    }
    if (tid < EMBED) btl[tid] = bt[tid];
    __syncthreads();

    int mi = w & 1, ni = w >> 1;
    int m0 = mi * 16, n0 = ni * 16;
    int hi = lane >> 4;
    int ncol = n0 + (lane & 15);
    float btn = btl[ncol];

    v16bf bf[4];                                       // resident across all steps
#pragma unroll
    for (int kc = 0; kc < 4; ++kc) bf[kc] = ld_b_frag(wt, n0, KSTR, kc * 32, lane);

    for (int t = 0; t < NSTEP; ++t) {
        // emit current z as bf16 rows (row = b*NSTEP + t)
        {
            int b  = tid >> 4;
            int e0 = (tid & 15) * 8;
            uint4 v = *(const uint4*)(zb + b * KSTR + e0);
            *(uint4*)(zs + ((size_t)(b * NSTEP + t)) * EMBED + e0) = v;
        }
        v8f acc;
#pragma unroll
        for (int j = 0; j < 8; ++j)
            acc[j] = zf[(m0 + hi * 8 + j) * ZSTRF + ncol] + btn;   // C = z + bt
        v16bf af[4];
#pragma unroll
        for (int kc = 0; kc < 4; ++kc) af[kc] = ld_a_frag(zb, m0, KSTR, kc * 32, lane);
        __syncthreads();                                // all reads of z done
#pragma unroll
        for (int kc = 0; kc < 4; ++kc)
            acc = __builtin_amdgcn_wmma_f32_16x16x32_bf16(false, af[kc], false, bf[kc],
                                                          (short)0, acc, false, false);
#pragma unroll
        for (int j = 0; j < 8; ++j) {
            float v = fast_tanh(acc[j]);
            int m = m0 + hi * 8 + j;
            zf[m * ZSTRF + ncol] = v;
            zb[m * KSTR + ncol]  = f2bf(v);
        }
        __syncthreads();
    }
}

// ---------------------------------------------------------------------------
// Kernel 2: logits GEMM (bf16 WMMA, f32 acc, C-init = bv) fused with online
// per-row (max, sum-exp) over an 832-wide vocab chunk. Grid 64 x 64.
// B tiles double-buffered via GLOBAL_LOAD_ASYNC_TO_LDS_B128 (ASYNCcnt).
// ---------------------------------------------------------------------------
__global__ __launch_bounds__(256) void k_logits(const unsigned short* __restrict__ zsb,
                                                const unsigned short* __restrict__ wvb,
                                                const float* __restrict__ bv,
                                                float* __restrict__ partM,
                                                float* __restrict__ partS) {
    __shared__ unsigned short As[64 * KSTR];
    __shared__ unsigned short Bs0[64 * KSTR];
    __shared__ unsigned short Bs1[64 * KSTR];
    __shared__ float pmax[4][64];
    __shared__ float psum[4][64];
    __shared__ float runM[64];
    __shared__ float runS[64];

    int tid = threadIdx.x, lane = tid & 31, w = tid >> 5;
    int wm = w >> 2, wn = w & 3;
    int mtile = blockIdx.x * 64;
    int vbase = blockIdx.y * CHUNK;

    if (vbase >= GDIM) {                    // chunks 61..63: fully out of vocab
        if (tid < 64) {
            size_t row = (size_t)mtile + tid;
            partM[row * NCHUNK + blockIdx.y] = -__builtin_inff();
            partS[row * NCHUNK + blockIdx.y] = 0.f;
        }
        return;
    }

    int hi = lane >> 4;
    int nloc = wn * 16 + (lane & 15);

    // async-DMA the A tile and the first B tile into LDS
    load_tile_async(As,  zsb + (size_t)mtile * EMBED, tid);
    load_tile_async(Bs0, wvb + (size_t)vbase * EMBED, tid);
    if (tid < 64) { runM[tid] = -__builtin_inff(); runS[tid] = 0.f; }
    wait_async0();
    __syncthreads();

    for (int vs = 0; vs < 13; ++vs) {
        unsigned short* cur = (vs & 1) ? Bs1 : Bs0;
        unsigned short* nxt = (vs & 1) ? Bs0 : Bs1;
        int v0 = vbase + vs * 64;
        if (vs + 1 < 13)                    // prefetch next tile while computing
            load_tile_async(nxt, wvb + (size_t)(v0 + 64) * EMBED, tid);

        int  nglob = v0 + nloc;
        bool valid = nglob < GDIM;
        float cinit = valid ? bv[nglob] : 0.f;
        v8f acc0, acc1;
#pragma unroll
        for (int j = 0; j < 8; ++j) { acc0[j] = cinit; acc1[j] = cinit; }
#pragma unroll
        for (int kc = 0; kc < 4; ++kc) {
            v16bf a0 = ld_a_frag(As,  wm * 32,      KSTR, kc * 32, lane);
            v16bf a1 = ld_a_frag(As,  wm * 32 + 16, KSTR, kc * 32, lane);
            v16bf bb = ld_b_frag(cur, wn * 16,      KSTR, kc * 32, lane);
            acc0 = __builtin_amdgcn_wmma_f32_16x16x32_bf16(false, a0, false, bb,
                                                           (short)0, acc0, false, false);
            acc1 = __builtin_amdgcn_wmma_f32_16x16x32_bf16(false, a1, false, bb,
                                                           (short)0, acc1, false, false);
        }

        float rmax0[8], rsum0[8], rmax1[8], rsum1[8];
#pragma unroll
        for (int j = 0; j < 8; ++j) {
            float v0v = valid ? acc0[j] : -__builtin_inff();
            float v1v = valid ? acc1[j] : -__builtin_inff();
            float m0v = v0v, m1v = v1v;
#pragma unroll
            for (int s = 1; s < 16; s <<= 1) {          // within 16-lane half
                m0v = fmaxf(m0v, __shfl_xor(m0v, s, 32));
                m1v = fmaxf(m1v, __shfl_xor(m1v, s, 32));
            }
            float e0 = (m0v == -__builtin_inff()) ? 0.f : __expf(v0v - m0v);
            float e1 = (m1v == -__builtin_inff()) ? 0.f : __expf(v1v - m1v);
#pragma unroll
            for (int s = 1; s < 16; s <<= 1) {
                e0 += __shfl_xor(e0, s, 32);
                e1 += __shfl_xor(e1, s, 32);
            }
            rmax0[j] = m0v; rsum0[j] = e0; rmax1[j] = m1v; rsum1[j] = e1;
        }
        if ((lane & 15) == 0) {
            int rb = wm * 32 + hi * 8;
#pragma unroll
            for (int j = 0; j < 8; ++j) {
                pmax[wn][rb + j]      = rmax0[j]; psum[wn][rb + j]      = rsum0[j];
                pmax[wn][rb + 16 + j] = rmax1[j]; psum[wn][rb + 16 + j] = rsum1[j];
            }
        }
        __syncthreads();
        if (tid < 64) {                                 // online merge, 1 thread/row
            float M = runM[tid], S = runS[tid];
            float m4 = fmaxf(fmaxf(pmax[0][tid], pmax[1][tid]),
                             fmaxf(pmax[2][tid], pmax[3][tid]));
            float nM = fmaxf(M, m4);
            float nS = (S > 0.f) ? S * __expf(M - nM) : 0.f;
#pragma unroll
            for (int q = 0; q < 4; ++q) {
                float ps = psum[q][tid];
                if (ps > 0.f) nS += ps * __expf(pmax[q][tid] - nM);
            }
            runM[tid] = nM; runS[tid] = nS;
        }
        wait_async0();                  // next B tile landed in LDS
        __syncthreads();
    }
    if (tid < 64) {
        size_t row = (size_t)mtile + tid;
        partM[row * NCHUNK + blockIdx.y] = runM[tid];
        partS[row * NCHUNK + blockIdx.y] = runS[tid];
    }
}

// ---------------------------------------------------------------------------
// Kernel 3: merge 64 chunk partials per row + recompute logit[y]; one wave/row
// ---------------------------------------------------------------------------
__global__ __launch_bounds__(256) void k_final(const float* __restrict__ partM,
                                               const float* __restrict__ partS,
                                               const unsigned short* __restrict__ zsb,
                                               const unsigned short* __restrict__ wvb,
                                               const float* __restrict__ bv,
                                               const int* __restrict__ y,
                                               float* __restrict__ out) {
    int lane = threadIdx.x & 31;
    int row  = blockIdx.x * 8 + (threadIdx.x >> 5);
    float m0 = partM[(size_t)row * NCHUNK + lane];
    float m1 = partM[(size_t)row * NCHUNK + 32 + lane];
    float s0 = partS[(size_t)row * NCHUNK + lane];
    float s1 = partS[(size_t)row * NCHUNK + 32 + lane];
    float M = fmaxf(m0, m1);
    for (int s = 1; s < 32; s <<= 1) M = fmaxf(M, __shfl_xor(M, s, 32));
    float S = 0.f;
    if (s0 > 0.f) S += s0 * __expf(m0 - M);
    if (s1 > 0.f) S += s1 * __expf(m1 - M);
    for (int s = 1; s < 32; s <<= 1) S += __shfl_xor(S, s, 32);

    int yv = y[row];
    float d = 0.f;
#pragma unroll
    for (int q = 0; q < 4; ++q) {
        int k = lane * 4 + q;
        d += bf2f(zsb[(size_t)row * EMBED + k]) * bf2f(wvb[(size_t)yv * EMBED + k]);
    }
    for (int s = 1; s < 32; s <<= 1) d += __shfl_xor(d, s, 32);
    if (lane == 0)
        out[row] = (d + bv[yv]) - (M + __logf(S));
}

// ---------------------------------------------------------------------------
extern "C" void kernel_launch(void* const* d_in, const int* in_sizes, int n_in,
                              void* d_out, int out_size, void* d_ws, size_t ws_size,
                              hipStream_t stream) {
    const int*   zi     = (const int*)d_in[0];
    const int*   y      = (const int*)d_in[1];
    const float* latent = (const float*)d_in[2];
    const float* Wt     = (const float*)d_in[3];
    const float* bt     = (const float*)d_in[4];
    const float* Wv     = (const float*)d_in[5];
    const float* bv     = (const float*)d_in[6];
    float*       out    = (float*)d_out;

    char* ws = (char*)d_ws;
    unsigned short* zsb  = (unsigned short*)ws;                          // 1 MiB
    unsigned short* wvb  = (unsigned short*)(ws + (size_t)1048576);      // 13 MiB
    float*          pM   = (float*)(ws + (size_t)1048576 + 13631488);    // 1 MiB
    float*          pS   = pM + (size_t)MROWS * NCHUNK;                  // 1 MiB

    k_cvt   <<<(VPAD * EMBED) / 256, 256, 0, stream>>>(Wv, wvb);
    k_rnn   <<<1, 512, 0, stream>>>(zi, latent, Wt, bt, zsb);
    k_logits<<<dim3(MROWS / 64, NCHUNK), 256, 0, stream>>>(zsb, wvb, bv, pM, pS);
    k_final <<<MROWS / 8, 256, 0, stream>>>(pM, pS, zsb, wvb, bv, y, out);
}
